// GCNModel_84628035600885
// MI455X (gfx1250) — matrix operand order repaired and checked by reading, not compiled, not run on hardware
//
#include <hip/hip_runtime.h>
#include <hip/hip_bf16.h>

// CDNA5 wave32 WMMA vector types
typedef __attribute__((ext_vector_type(16))) __bf16 v16bf;
typedef __attribute__((ext_vector_type(8)))  __bf16 v8bf;
typedef __attribute__((ext_vector_type(4)))  __bf16 v4bf;
typedef __attribute__((ext_vector_type(8)))  float  v8f;

#define HID 64

// ---------------------------------------------------------------------------
// degree / normalization precompute
// ---------------------------------------------------------------------------
__global__ __launch_bounds__(256) void k_deg_init(float* __restrict__ deg, int n) {
  int i = blockIdx.x * 256 + threadIdx.x;
  if (i < n) deg[i] = 1.0f;  // self-loop
}

__global__ __launch_bounds__(256) void k_deg_accum(const int* __restrict__ dst,
                                                   float* __restrict__ deg, int E) {
  int e = blockIdx.x * 256 + threadIdx.x;
  if (e < E) atomicAdd(&deg[dst[e]], 1.0f);
}

__global__ __launch_bounds__(256) void k_dinv(float* __restrict__ deg, int n) {
  int i = blockIdx.x * 256 + threadIdx.x;
  if (i < n) {
    float d = deg[i];
    deg[i] = (d > 0.0f) ? rsqrtf(d) : 0.0f;
  }
}

__global__ __launch_bounds__(256) void k_norm(const int* __restrict__ src,
                                              const int* __restrict__ dst,
                                              const float* __restrict__ dinv,
                                              float* __restrict__ norm, int E) {
  int e = blockIdx.x * 256 + threadIdx.x;
  if (e < E) norm[e] = dinv[src[e]] * dinv[dst[e]];
}

// ---------------------------------------------------------------------------
// fp32 -> bf16 row-major convert (4 elems / thread, b128 in / b64 out)
// ---------------------------------------------------------------------------
__global__ __launch_bounds__(256) void k_cvt4(const float4* __restrict__ s,
                                              v4bf* __restrict__ d, int n4) {
  int t = blockIdx.x * 256 + threadIdx.x;
  if (t >= n4) return;
  const float4 v = s[t];
  v4bf o;
  o[0] = (__bf16)v.x; o[1] = (__bf16)v.y; o[2] = (__bf16)v.z; o[3] = (__bf16)v.w;
  d[t] = o;
}

// ---------------------------------------------------------------------------
// Pack W[KTOT x 64] (f32, row-major) into per-lane WMMA B-fragment order:
//   Wp[(ks*4 + colTile)*512 + lane*16 + j] = bf16( W[(ks*32 + 8*(lane>>4) + (j&7) + 16*(j>>3))*64
//                                                   + colTile*16 + (lane&15)] )
// so each lane's 16-elem fragment is one contiguous 32-byte chunk.
// ---------------------------------------------------------------------------
__global__ __launch_bounds__(256) void k_packW(const float* __restrict__ W,
                                               __bf16* __restrict__ Wp, int total) {
  int t = blockIdx.x * 256 + threadIdx.x;
  if (t >= total) return;
  const int j       = t & 15;
  const int lane    = (t >> 4) & 31;
  const int tile    = t >> 9;           // ks*4 + colTile
  const int colTile = tile & 3;
  const int ks      = tile >> 2;
  const int g = lane >> 4;
  const int m = lane & 15;
  const int k = ks * 32 + 8 * g + (j & 7) + 16 * (j >> 3);
  Wp[t] = (__bf16)W[k * HID + colTile * 16 + m];
}

// ---------------------------------------------------------------------------
// Dense transform C[N x 64] = A[N x KTOT](bf16) @ Wp(pre-packed bf16)
//   8 waves / block; wave w: rowTile = w>>2 (of 2), colTile = w&3 (of 4)
//   -> block covers 32 rows x 64 cols. EXEC all-1s inside WMMA waves.
// Inner loop per K-step: 2x b128 A loads + 32B B load + 1x v_wmma. No converts.
// ---------------------------------------------------------------------------
template <int KTOT>
__global__ __launch_bounds__(256) void k_gemm_wmma(const __bf16* __restrict__ A,
                                                   const __bf16* __restrict__ Wp,
                                                   float* __restrict__ C, int nrows) {
  const int lane = threadIdx.x & 31;
  const int wave = threadIdx.x >> 5;
  const int rowBase = blockIdx.x * 32 + (wave >> 2) * 16;
  const int colTile = wave & 3;
  if (rowBase + 16 > nrows) return;  // wave-uniform: EXEC stays all-1s

  const int g = lane >> 4;
  const int m = lane & 15;

  const __bf16* __restrict__ arow = A + (size_t)(rowBase + m) * KTOT;

  v8f acc = {};
#pragma unroll
  for (int ks = 0; ks < KTOT / 32; ++ks) {
    const int kb = ks * 32;
    // A fragment: two contiguous 8-elem (16B, aligned) runs
    const v8bf a0 = *(const v8bf*)(arow + kb + 8 * g);
    const v8bf a1 = *(const v8bf*)(arow + kb + 16 + 8 * g);
    const v16bf a = __builtin_shufflevector(a0, a1, 0, 1, 2, 3, 4, 5, 6, 7,
                                            8, 9, 10, 11, 12, 13, 14, 15);
    // B fragment: one contiguous 32B chunk per lane
    const v16bf b = *(const v16bf*)(Wp + (size_t)(ks * 4 + colTile) * 512 + lane * 16);
    // (neg_a, A, neg_b, B, c_mod, C, reuse_a, reuse_b)
    acc = __builtin_amdgcn_wmma_f32_16x16x32_bf16(false, a, false, b,
                                                  (short)0, acc, false, false);
  }

  float* __restrict__ crow = C + (size_t)rowBase * HID + colTile * 16 + m;
#pragma unroll
  for (int r = 0; r < 8; ++r) {
    crow[(size_t)(r + 8 * g) * HID] = acc[r];
  }
}

// ---------------------------------------------------------------------------
// Edge scatter: out[dst] += tmp[src] * norm   (16 threads/edge, float4 each)
// Accumulator table is 25.6 MB -> L2-resident atomics.
// ---------------------------------------------------------------------------
__global__ __launch_bounds__(256) void k_scatter(const int* __restrict__ src,
                                                 const int* __restrict__ dst,
                                                 const float* __restrict__ norm,
                                                 const float* __restrict__ tmp,
                                                 float* __restrict__ out, int E) {
  const long long t = (long long)blockIdx.x * 256 + threadIdx.x;
  const int e = (int)(t >> 4);
  if (e >= E) return;
  const int q = (int)(t & 15);

  const int s = src[e];
  const int d = dst[e];
  const float w = norm[e];

  const float4 v = *(const float4*)(tmp + (size_t)s * HID + q * 4);
  float* o = out + (size_t)d * HID + q * 4;
  atomicAdd(o + 0, v.x * w);
  atomicAdd(o + 1, v.y * w);
  atomicAdd(o + 2, v.z * w);
  atomicAdd(o + 3, v.w * w);
}

// ---------------------------------------------------------------------------
// Epilogue: h = relu(out + tmp * dinv^2 (self-loop) + bias)
// Writes f32 h in place AND a bf16 copy (next layer's WMMA A operand).
// ---------------------------------------------------------------------------
__global__ __launch_bounds__(256) void k_finish(float* __restrict__ out,
                                                const float* __restrict__ tmp,
                                                const float* __restrict__ dinv,
                                                const float* __restrict__ bias,
                                                __bf16* __restrict__ hbf, int n) {
  const int t = blockIdx.x * 256 + threadIdx.x;
  if (t >= n * HID) return;
  const int i = t >> 6;
  const int c = t & (HID - 1);
  const float di = dinv[i];
  const float v = fmaxf(out[t] + tmp[t] * di * di + bias[c], 0.0f);
  out[t] = v;
  hbf[t] = (__bf16)v;
}

// ---------------------------------------------------------------------------
// Global mean pool accumulation + head
// ---------------------------------------------------------------------------
__global__ __launch_bounds__(256) void k_pool(const float* __restrict__ h,
                                              const int* __restrict__ batch,
                                              float* __restrict__ pooled,
                                              float* __restrict__ cnt, int n) {
  const int t = blockIdx.x * 256 + threadIdx.x;
  if (t >= n * HID) return;
  const int i = t >> 6;
  const int c = t & (HID - 1);
  const int gi = batch[i];
  atomicAdd(&pooled[gi * HID + c], h[t]);
  if (c == 0) atomicAdd(&cnt[gi], 1.0f);
}

__global__ __launch_bounds__(64) void k_final(const float* __restrict__ pooled,
                                              const float* __restrict__ cnt,
                                              const float* __restrict__ Wl,
                                              const float* __restrict__ bl,
                                              float* __restrict__ out, int G) {
  const int gi = threadIdx.x;
  if (gi >= G) return;
  float s = 0.0f;
#pragma unroll
  for (int c = 0; c < HID; ++c) s += pooled[gi * HID + c] * Wl[c];
  out[gi] = s / fmaxf(cnt[gi], 1.0f) + bl[0];
}

// ---------------------------------------------------------------------------
extern "C" void kernel_launch(void* const* d_in, const int* in_sizes, int n_in,
                              void* d_out, int out_size, void* d_ws, size_t ws_size,
                              hipStream_t stream) {
  (void)n_in; (void)ws_size;

  const float* x    = (const float*)d_in[0];
  const int*   ei   = (const int*)d_in[1];
  const int*   batch= (const int*)d_in[2];
  const float* W1   = (const float*)d_in[3];
  const float* b1   = (const float*)d_in[4];
  const float* W2   = (const float*)d_in[5];
  const float* b2   = (const float*)d_in[6];
  const float* W3   = (const float*)d_in[7];
  const float* b3   = (const float*)d_in[8];
  const float* Wl   = (const float*)d_in[9];
  const float* bl   = (const float*)d_in[10];
  float* out = (float*)d_out;

  const int N = in_sizes[2];          // 100000
  const int E = in_sizes[1] / 2;      // 3200000
  const int F = in_sizes[0] / N;      // 128
  const int G = out_size;             // 64

  const int* src = ei;
  const int* dst = ei + E;

  // workspace layout (bytes, 256B-aligned chunks)
  auto al = [](size_t v) { return (v + 255) & ~(size_t)255; };
  char* wsb = (char*)d_ws;
  size_t off = 0;
  float*  dinv   = (float*)(wsb + off);  off += al((size_t)N * 4);
  float*  norm   = (float*)(wsb + off);  off += al((size_t)E * 4);
  float*  bufA   = (float*)(wsb + off);  off += al((size_t)N * HID * 4);  // tmp = h @ W
  float*  bufB   = (float*)(wsb + off);  off += al((size_t)N * HID * 4);  // scattered out / h
  __bf16* xbf    = (__bf16*)(wsb + off); off += al((size_t)N * F * 2);    // bf16 x
  __bf16* hbf    = (__bf16*)(wsb + off); off += al((size_t)N * HID * 2);  // bf16 h
  __bf16* Wp     = (__bf16*)(wsb + off); off += al((size_t)F * HID * 2);  // packed weights
  float*  pooled = (float*)(wsb + off);                                   // G*HID then cnt[G]
  float*  cnt    = pooled + (size_t)G * HID;

  const int nBlk  = (N + 255) / 256;
  const int eBlk  = (E + 255) / 256;
  const int nfBlk = (N * HID + 255) / 256;
  const int sBlk  = (int)(((long long)E * 16 + 255) / 256);
  const int gBlk  = (N + 31) / 32;
  const size_t hbytes = (size_t)N * HID * sizeof(float);

  // degree / normalization (shared by all 3 layers)
  k_deg_init <<<nBlk, 256, 0, stream>>>(dinv, N);
  k_deg_accum<<<eBlk, 256, 0, stream>>>(dst, dinv, E);
  k_dinv     <<<nBlk, 256, 0, stream>>>(dinv, N);
  k_norm     <<<eBlk, 256, 0, stream>>>(src, dst, dinv, norm, E);

  // convert x to bf16 once per call
  k_cvt4<<<((N * F / 4) + 255) / 256, 256, 0, stream>>>((const float4*)x, (v4bf*)xbf, N * F / 4);

  // layer 1 (K = 128)
  k_packW<<<(F * HID + 255) / 256, 256, 0, stream>>>(W1, Wp, F * HID);
  k_gemm_wmma<128><<<gBlk, 256, 0, stream>>>(xbf, Wp, bufA, N);
  hipMemsetAsync(bufB, 0, hbytes, stream);
  k_scatter<<<sBlk, 256, 0, stream>>>(src, dst, norm, bufA, bufB, E);
  k_finish <<<nfBlk, 256, 0, stream>>>(bufB, bufA, dinv, b1, hbf, N);

  // layer 2 (K = 64)
  k_packW<<<(HID * HID + 255) / 256, 256, 0, stream>>>(W2, Wp, HID * HID);
  k_gemm_wmma<64><<<gBlk, 256, 0, stream>>>(hbf, Wp, bufA, N);
  hipMemsetAsync(bufB, 0, hbytes, stream);
  k_scatter<<<sBlk, 256, 0, stream>>>(src, dst, norm, bufA, bufB, E);
  k_finish <<<nfBlk, 256, 0, stream>>>(bufB, bufA, dinv, b2, hbf, N);

  // layer 3 (K = 64)
  k_packW<<<(HID * HID + 255) / 256, 256, 0, stream>>>(W3, Wp, HID * HID);
  k_gemm_wmma<64><<<gBlk, 256, 0, stream>>>(hbf, Wp, bufA, N);
  hipMemsetAsync(bufB, 0, hbytes, stream);
  k_scatter<<<sBlk, 256, 0, stream>>>(src, dst, norm, bufA, bufB, E);
  k_finish <<<nfBlk, 256, 0, stream>>>(bufB, bufA, dinv, b3, hbf, N);

  // global mean pool + linear head
  hipMemsetAsync(pooled, 0, (size_t)(G * HID + G) * sizeof(float), stream);
  k_pool <<<nfBlk, 256, 0, stream>>>(bufB, batch, pooled, cnt, N);
  k_final<<<1, 64, 0, stream>>>(pooled, cnt, Wl, bl, out, G);
}
